// DecoderBlock_49709951484780
// MI455X (gfx1250) — compile-verified, matrix operand order checked by reading, not run on hardware
//
#include <hip/hip_runtime.h>

// ---------------------------------------------------------------- constants
#define S_LEN   2048
#define DIM_    512
#define L_LEN   10240      // 5*S
#define KDIM_   136        // PROJ + NPOS
#define K1      160        // KDIM padded to multiple of 32
#define HEADS_  4
#define HD      128        // head dim
#define KCONV   1536       // 3*DIM im2col K
#define PROJ_   128
#define NPOS_   8

// ---------------------------------------------------------------- types
typedef __bf16 bf16;
typedef __bf16  v16bf __attribute__((ext_vector_type(16)));
typedef float   v8f   __attribute__((ext_vector_type(8)));

struct __align__(16) Q16 { unsigned u0, u1, u2, u3; };   // 8 bf16 / 16 bytes
union FragA { v16bf v; Q16 q[2]; };

__device__ __forceinline__ bf16 f2bf(float f) {
  union { float f; unsigned u; } a; a.f = f;
  unsigned r = a.u + 0x7FFFu + ((a.u >> 16) & 1u);
  union { unsigned short s; bf16 b; } o; o.s = (unsigned short)(r >> 16);
  return o.b;
}

__device__ __forceinline__ v8f wmma_bf(const FragA& a, const FragA& b, v8f c) {
  return __builtin_amdgcn_wmma_f32_16x16x32_bf16(
      false, a.v, false, b.v, (short)0, c, false, false);
}

__device__ __forceinline__ float wred(float v) {
#pragma unroll
  for (int m = 16; m; m >>= 1) v += __shfl_xor(v, m, 32);
  return v;
}

// CDNA5 async DMA: global -> LDS, 16B per lane, tracked by ASYNCcnt.
__device__ __forceinline__ void async_g2l_b128(unsigned lds_off, const void* gptr) {
  asm volatile("global_load_async_to_lds_b128 %0, %1, off"
               :: "v"(lds_off), "v"((unsigned long long)gptr) : "memory");
}
__device__ __forceinline__ void wait_async0() {
  asm volatile("s_wait_asynccnt 0x0" ::: "memory");
}

// ---------------------------------------------------------------- converters
// src = [x | pos] padded to K1 cols, bf16
__global__ __launch_bounds__(256) void k_build_src(const float* __restrict__ x,
                                                   bf16* __restrict__ dst) {
  int idx = blockIdx.x * 256 + threadIdx.x;
  if (idx >= S_LEN * K1) return;
  int n = idx / K1, c = idx % K1;
  float v;
  if (c < PROJ_) {
    v = x[n * PROJ_ + c];
  } else if (c < KDIM_) {
    int e = c - PROJ_;
    v = (float)(n & ((2 << e) - 1)) / (float)(1 << e);
  } else {
    v = 0.f;
  }
  dst[idx] = f2bf(v);
}

// weight [N,Ks] f32 -> [N,Kd] bf16 zero padded
__global__ __launch_bounds__(256) void k_pad_convert(const float* __restrict__ w,
                                                     bf16* __restrict__ dst,
                                                     int N, int Ks, int Kd) {
  int idx = blockIdx.x * 256 + threadIdx.x;
  if (idx >= N * Kd) return;
  int n = idx / Kd, c = idx % Kd;
  dst[idx] = (c < Ks) ? f2bf(w[(size_t)n * Ks + c]) : f2bf(0.f);
}

// conv_w [3,512,512,3] -> per layer W2[o][kk*512+i] bf16
__global__ __launch_bounds__(256) void k_conv_w(const float* __restrict__ cw,
                                                bf16* __restrict__ dst) {
  int idx = blockIdx.x * 256 + threadIdx.x;
  const int total = 3 * DIM_ * DIM_ * 3;
  if (idx >= total) return;
  int kk = idx % 3;
  int i  = (idx / 3) % DIM_;
  int o  = (idx / (3 * DIM_)) % DIM_;
  int l  = idx / (3 * DIM_ * DIM_);
  dst[((size_t)(l * DIM_ + o)) * KCONV + kk * DIM_ + i] = f2bf(cw[idx]);
}

// im2col: out[t][dt*512+c] = in[t-1+dt][c] (zeros at edges), 8 elems/thread
__global__ __launch_bounds__(256) void k_im2col(const bf16* __restrict__ in,
                                                bf16* __restrict__ out) {
  int idx = blockIdx.x * 256 + threadIdx.x;
  const int total = L_LEN * (KCONV / 8);
  if (idx >= total) return;
  int t  = idx / (KCONV / 8);
  int c8 = (idx % (KCONV / 8)) * 8;
  int dt = c8 >> 9;          // /512
  int cc = c8 & 511;
  int sr = t + dt - 1;
  Q16 z; z.u0 = z.u1 = z.u2 = z.u3 = 0u;
  Q16 val = (sr >= 0 && sr < L_LEN)
                ? *(const Q16*)(in + (size_t)sr * DIM_ + cc) : z;
  *(Q16*)(out + (size_t)t * KCONV + c8) = val;
}

__global__ __launch_bounds__(256) void k_softplus(const float* __restrict__ h,
                                                  float* __restrict__ outf,
                                                  bf16* __restrict__ outbf, int n) {
  int i = blockIdx.x * 256 + threadIdx.x;
  if (i >= n) return;
  float x = h[i];
  float y = (x > 20.f) ? x : log1pf(__expf(x));
  outf[i]  = y;
  outbf[i] = f2bf(y);
}

// ---------------------------------------------------------------- LayerNorm
// y = LN(a [+ bpre]) [+ post];  one wave per row of 512
__global__ __launch_bounds__(256) void k_ln(const float* __restrict__ a,
                                            const float* __restrict__ bpre,
                                            const float* __restrict__ post,
                                            float* __restrict__ outf,
                                            bf16* __restrict__ outbf) {
  int wave = threadIdx.x >> 5, lane = threadIdx.x & 31;
  int row = blockIdx.x * 8 + wave;
  size_t base = (size_t)row * DIM_;
  float v[16];
  float s = 0.f, s2 = 0.f;
#pragma unroll
  for (int jj = 0; jj < 4; jj++) {
    int e0 = (lane + jj * 32) * 4;
#pragma unroll
    for (int c = 0; c < 4; c++) {
      float x = a[base + e0 + c];
      if (bpre) x += bpre[base + e0 + c];
      v[jj * 4 + c] = x; s += x; s2 += x * x;
    }
  }
  s = wred(s); s2 = wred(s2);
  float mean = s * (1.f / DIM_);
  float var  = s2 * (1.f / DIM_) - mean * mean;
  float rstd = rsqrtf(var + 1e-5f);
#pragma unroll
  for (int jj = 0; jj < 4; jj++) {
    int e0 = (lane + jj * 32) * 4;
#pragma unroll
    for (int c = 0; c < 4; c++) {
      float y = (v[jj * 4 + c] - mean) * rstd;
      if (post) y += post[base + e0 + c];
      outf[base + e0 + c] = y;
      if (outbf) outbf[base + e0 + c] = f2bf(y);
    }
  }
}

// ---------------------------------------------------------------- WMMA GEMM
// C[M,N] = A[M,K](bf16) @ B[N,K]^T(bf16) + bias[n]
// block tile 128x64, 8 waves, each wave 32x32 (2x2 WMMA 16x16x32).
// A/B tiles staged into LDS via CDNA5 async DMA (global_load_async_to_lds_b128,
// ASYNCcnt), fragments consumed with ds_load_b128.
// requires M%128==0, N%64==0, K%32==0, lda=ldb=K
__global__ __launch_bounds__(256) void k_gemm(const bf16* __restrict__ A,
                                              const bf16* __restrict__ B,
                                              const float* __restrict__ bias,
                                              float* __restrict__ Cf,
                                              bf16* __restrict__ Cbf,
                                              int M, int N, int K) {
  __shared__ bf16 As[128 * 40];
  __shared__ bf16 Bs[64 * 40];
  const int tid = threadIdx.x, lane = tid & 31, wave = tid >> 5;
  const int wm = (wave & 3) * 32, wn = (wave >> 2) * 32;
  const int m0 = blockIdx.y * 128, n0 = blockIdx.x * 64;
  const int hi8 = (lane >= 16) ? 8 : 0;
  const int mr = lane & 15;

  // staging tasks: 16B (8 bf16) chunks
  const int ar0 = tid >> 2;            // A rows 0..63
  const int ar1 = 64 + ar0;            // A rows 64..127
  const int ac  = (tid & 3) * 8;       // chunk within the 32-wide K slice
  const int br  = tid >> 2;            // B rows 0..63

  const unsigned la0 = (unsigned)(uintptr_t)&As[ar0 * 40 + ac];
  const unsigned la1 = (unsigned)(uintptr_t)&As[ar1 * 40 + ac];
  const unsigned lb  = (unsigned)(uintptr_t)&Bs[br * 40 + ac];
  const bf16* ga0 = A + (size_t)(m0 + ar0) * K + ac;
  const bf16* ga1 = A + (size_t)(m0 + ar1) * K + ac;
  const bf16* gb  = B + (size_t)(n0 + br) * K + ac;

  v8f c00 = {}, c01 = {}, c10 = {}, c11 = {};

  for (int kk = 0; kk < K; kk += 32) {
    // async DMA the 128x32 A tile and 64x32 B tile straight into LDS
    async_g2l_b128(la0, ga0 + kk);
    async_g2l_b128(la1, ga1 + kk);
    async_g2l_b128(lb,  gb  + kk);
    wait_async0();
    __syncthreads();

    FragA a0, a1, b0, b1;
    a0.q[0] = *(const Q16*)&As[(wm + mr) * 40 + hi8];
    a0.q[1] = *(const Q16*)&As[(wm + mr) * 40 + 16 + hi8];
    a1.q[0] = *(const Q16*)&As[(wm + 16 + mr) * 40 + hi8];
    a1.q[1] = *(const Q16*)&As[(wm + 16 + mr) * 40 + 16 + hi8];
    b0.q[0] = *(const Q16*)&Bs[(wn + mr) * 40 + hi8];
    b0.q[1] = *(const Q16*)&Bs[(wn + mr) * 40 + 16 + hi8];
    b1.q[0] = *(const Q16*)&Bs[(wn + 16 + mr) * 40 + hi8];
    b1.q[1] = *(const Q16*)&Bs[(wn + 16 + mr) * 40 + 16 + hi8];
    c00 = wmma_bf(a0, b0, c00);
    c01 = wmma_bf(a0, b1, c01);
    c10 = wmma_bf(a1, b0, c10);
    c11 = wmma_bf(a1, b1, c11);
    __syncthreads();
  }

#pragma unroll
  for (int i = 0; i < 2; i++)
#pragma unroll
    for (int j = 0; j < 2; j++) {
      v8f c = (i == 0) ? ((j == 0) ? c00 : c01) : ((j == 0) ? c10 : c11);
      int nn = n0 + wn + j * 16 + mr;
      float bz = bias ? bias[nn] : 0.f;
#pragma unroll
      for (int r = 0; r < 8; r++) {
        int mm = m0 + wm + i * 16 + r + hi8;
        float val = c[r] + bz;
        if (Cf)  Cf[(size_t)mm * N + nn]  = val;
        if (Cbf) Cbf[(size_t)mm * N + nn] = f2bf(val);
      }
    }
}

// ---------------------------------------------------------------- attention
// banded MHA: block = (128 queries, 1 head), 8 waves -> 16 queries/wave.
// shared 64-col aligned source window (covers the band for all 128 queries).
__global__ __launch_bounds__(256) void k_attn(const bf16* __restrict__ q,
                                              const bf16* __restrict__ kmat,
                                              const bf16* __restrict__ vmat,
                                              bf16* __restrict__ o) {
  extern __shared__ char smem[];
  bf16*  kwin   = (bf16*)smem;                  // [64][128]
  bf16*  vtw    = kwin + 64 * 128;              // [128][64] (transposed V)
  float* score  = (float*)(vtw + 128 * 64);     // [8][16*64]
  bf16*  attnp  = (bf16*)(score + 8 * 16 * 64); // [8][16*64]
  float* rowinv = (float*)(attnp + 8 * 16 * 64);// [8][16]

  const int tid = threadIdx.x, lane = tid & 31, wave = tid >> 5;
  const int h = blockIdx.y;
  const int l0_wg = blockIdx.x * 128;
  int imin = l0_wg - 68;
  imin = (imin > 0) ? (imin + 4) / 5 : 0;       // ceil((l0-68)/5), clamped
  const int ib = imin & ~7;                     // 8-aligned window base

  { // stage K window: kwin[r][d]
    int r = tid >> 2, cq = (tid & 3) * 32;
    int src = min(ib + r, S_LEN - 1);
    const Q16* gp = (const Q16*)(kmat + (size_t)src * DIM_ + h * HD + cq);
    Q16* lp = (Q16*)&kwin[r * 128 + cq];
#pragma unroll
    for (int t = 0; t < 4; t++) lp[t] = gp[t];
  }
  // stage V window transposed: vtw[d][r]
#pragma unroll
  for (int tt = 0; tt < 4; tt++) {
    int task = tid + tt * 256;
    int r = task & 63, dc = task >> 6;
    int src = min(ib + r, S_LEN - 1);
    const bf16* vp = vmat + (size_t)src * DIM_ + h * HD + dc * 8;
#pragma unroll
    for (int jj = 0; jj < 8; jj++) vtw[(dc * 8 + jj) * 64 + r] = vp[jj];
  }
  __syncthreads();

  const int l0 = l0_wg + wave * 16;
  float* sc   = score + wave * 16 * 64;
  bf16*  at   = attnp + wave * 16 * 64;
  float* rinv = rowinv + wave * 16;
  const int mr = lane & 15;
  const int ko = (lane >= 16) ? 8 : 0;
  const int hi8 = ko;

  // scores: 16x64 tile, K = 128 (4 chunks of 32)
  v8f c[4] = {};
#pragma unroll
  for (int kc = 0; kc < 4; kc++) {
    FragA a;
    const bf16* qp = q + (size_t)(l0 + mr) * DIM_ + h * HD + kc * 32 + ko;
    a.q[0] = *(const Q16*)qp;
    a.q[1] = *(const Q16*)(qp + 16);
#pragma unroll
    for (int jt = 0; jt < 4; jt++) {
      FragA b;
      const bf16* kp = &kwin[(jt * 16 + mr) * 128 + kc * 32 + ko];
      b.q[0] = *(const Q16*)kp;
      b.q[1] = *(const Q16*)(kp + 16);
      c[jt] = wmma_bf(a, b, c[jt]);
    }
  }
  const float scale = 0.088388347648318447f;  // 1/sqrt(128)
#pragma unroll
  for (int jt = 0; jt < 4; jt++)
#pragma unroll
    for (int r = 0; r < 8; r++)
      sc[(r + hi8) * 64 + jt * 16 + mr] = c[jt][r] * scale;

  // softmax over the band (lanes 0..15, one row each)
  if (lane < 16) {
    int m = lane, l = l0 + m;
    float mx = -1e30f;
    for (int j = 0; j < 64; j++) {
      int i = ib + j;
      bool ok = (i < S_LEN) && (l >= 5 * i - 64) && (l < 5 * i + 69);
      float s = ok ? sc[m * 64 + j] : -1e30f;
      mx = fmaxf(mx, s);
    }
    float sum = 0.f;
    for (int j = 0; j < 64; j++) {
      int i = ib + j;
      bool ok = (i < S_LEN) && (l >= 5 * i - 64) && (l < 5 * i + 69);
      float s = ok ? sc[m * 64 + j] : -1e30f;
      float e = __expf(s - mx);
      sum += e;
      at[m * 64 + j] = f2bf(e);          // unnormalized; 1/sum folded at output
    }
    rinv[m] = 1.f / fmaxf(sum, 1e-30f);
  }

  // o = attn(16x64) @ v(64x128), normalize rows at writeout
  for (int nt = 0; nt < 8; nt++) {
    v8f acc = {};
#pragma unroll
    for (int kc = 0; kc < 2; kc++) {
      FragA a, b;
      const bf16* ap = &at[mr * 64 + kc * 32 + ko];
      a.q[0] = *(const Q16*)ap; a.q[1] = *(const Q16*)(ap + 16);
      const bf16* bp = &vtw[(nt * 16 + mr) * 64 + kc * 32 + ko];
      b.q[0] = *(const Q16*)bp; b.q[1] = *(const Q16*)(bp + 16);
      acc = wmma_bf(a, b, acc);
    }
#pragma unroll
    for (int r = 0; r < 8; r++) {
      int m = r + hi8;
      o[(size_t)(l0 + m) * DIM_ + h * HD + nt * 16 + mr] =
          f2bf(acc[r] * rinv[m]);
    }
  }
}

// ---------------------------------------------------------------- launcher
extern "C" void kernel_launch(void* const* d_in, const int* in_sizes, int n_in,
                              void* d_out, int out_size, void* d_ws, size_t ws_size,
                              hipStream_t stream) {
  (void)in_sizes; (void)n_in; (void)out_size; (void)ws_size;
  const float* x        = (const float*)d_in[0];
  const float* residual = (const float*)d_in[1];
  const float* proj_w   = (const float*)d_in[2];
  const float* proj_b   = (const float*)d_in[3];
  const float* q_w      = (const float*)d_in[4];
  const float* k_w      = (const float*)d_in[5];
  const float* v_w      = (const float*)d_in[6];
  const float* in_b     = (const float*)d_in[7];
  const float* out_w    = (const float*)d_in[8];
  const float* out_b    = (const float*)d_in[9];
  const float* conv_w   = (const float*)d_in[10];
  const float* conv_b   = (const float*)d_in[11];
  const float* skip_w   = (const float*)d_in[12];
  const float* skip_b   = (const float*)d_in[13];
  float* outp = (float*)d_out;

  // workspace carving
  char* wp = (char*)d_ws;
  auto alloc = [&](size_t bytes) -> void* {
    void* r = (void*)wp;
    wp += (bytes + 255) & ~(size_t)255;
    return r;
  };
  bf16* src_bf = (bf16*)alloc((size_t)S_LEN * K1 * 2);
  bf16* pw_bf  = (bf16*)alloc((size_t)2560 * K1 * 2);
  bf16* qw_bf  = (bf16*)alloc((size_t)DIM_ * DIM_ * 2);
  bf16* kw_bf  = (bf16*)alloc((size_t)DIM_ * K1 * 2);
  bf16* vw_bf  = (bf16*)alloc((size_t)DIM_ * K1 * 2);
  bf16* ow_bf  = (bf16*)alloc((size_t)DIM_ * DIM_ * 2);
  bf16* sw_bf  = (bf16*)alloc((size_t)DIM_ * DIM_ * 2);
  bf16* cw_bf  = (bf16*)alloc((size_t)3 * DIM_ * KCONV * 2);
  bf16* tgt_bf = (bf16*)alloc((size_t)L_LEN * DIM_ * 2);
  bf16* q_bf   = (bf16*)alloc((size_t)L_LEN * DIM_ * 2);
  bf16* k_bf   = (bf16*)alloc((size_t)S_LEN * DIM_ * 2);
  bf16* v_bf   = (bf16*)alloc((size_t)S_LEN * DIM_ * 2);
  bf16* o_bf   = (bf16*)alloc((size_t)L_LEN * DIM_ * 2);
  bf16* cnn_bf = (bf16*)alloc((size_t)L_LEN * DIM_ * 2);
  bf16* act_bf = (bf16*)alloc((size_t)L_LEN * DIM_ * 2);
  bf16* col_bf = (bf16*)alloc((size_t)L_LEN * KCONV * 2);
  float* attn_f = (float*)alloc((size_t)L_LEN * DIM_ * 4);
  float* cnn_f  = (float*)alloc((size_t)L_LEN * DIM_ * 4);
  float* h_f    = (float*)alloc((size_t)L_LEN * DIM_ * 4);
  float* act_f  = (float*)alloc((size_t)L_LEN * DIM_ * 4);
  float* skip_f = (float*)alloc((size_t)L_LEN * DIM_ * 4);

  const int NT = L_LEN * DIM_;  // 5,242,880
  auto cdiv = [](int a, int b) { return (a + b - 1) / b; };

  // 1) build src + convert weights
  hipLaunchKernelGGL(k_build_src, dim3(cdiv(S_LEN * K1, 256)), dim3(256), 0, stream, x, src_bf);
  hipLaunchKernelGGL(k_pad_convert, dim3(cdiv(2560 * K1, 256)), dim3(256), 0, stream, proj_w, pw_bf, 2560, KDIM_, K1);
  hipLaunchKernelGGL(k_pad_convert, dim3(cdiv(DIM_ * DIM_, 256)), dim3(256), 0, stream, q_w, qw_bf, DIM_, DIM_, DIM_);
  hipLaunchKernelGGL(k_pad_convert, dim3(cdiv(DIM_ * K1, 256)), dim3(256), 0, stream, k_w, kw_bf, DIM_, KDIM_, K1);
  hipLaunchKernelGGL(k_pad_convert, dim3(cdiv(DIM_ * K1, 256)), dim3(256), 0, stream, v_w, vw_bf, DIM_, KDIM_, K1);
  hipLaunchKernelGGL(k_pad_convert, dim3(cdiv(DIM_ * DIM_, 256)), dim3(256), 0, stream, out_w, ow_bf, DIM_, DIM_, DIM_);
  hipLaunchKernelGGL(k_pad_convert, dim3(cdiv(DIM_ * DIM_, 256)), dim3(256), 0, stream, skip_w, sw_bf, DIM_, DIM_, DIM_);
  hipLaunchKernelGGL(k_conv_w, dim3(cdiv(3 * DIM_ * DIM_ * 3, 256)), dim3(256), 0, stream, conv_w, cw_bf);

  // 2) projections (WMMA GEMMs)
  hipLaunchKernelGGL(k_gemm, dim3(2560 / 64, S_LEN / 128), dim3(256), 0, stream,
                     src_bf, pw_bf, proj_b, (float*)nullptr, tgt_bf, S_LEN, 2560, K1);
  hipLaunchKernelGGL(k_gemm, dim3(DIM_ / 64, L_LEN / 128), dim3(256), 0, stream,
                     tgt_bf, qw_bf, in_b, (float*)nullptr, q_bf, L_LEN, DIM_, DIM_);
  hipLaunchKernelGGL(k_gemm, dim3(DIM_ / 64, S_LEN / 128), dim3(256), 0, stream,
                     src_bf, kw_bf, in_b + DIM_, (float*)nullptr, k_bf, S_LEN, DIM_, K1);
  hipLaunchKernelGGL(k_gemm, dim3(DIM_ / 64, S_LEN / 128), dim3(256), 0, stream,
                     src_bf, vw_bf, in_b + 2 * DIM_, (float*)nullptr, v_bf, S_LEN, DIM_, K1);

  // 3) banded attention (WMMA)
  {
    size_t smem = (size_t)64 * 128 * 2 + (size_t)128 * 64 * 2 +
                  (size_t)8 * 16 * 64 * 4 + (size_t)8 * 16 * 64 * 2 +
                  (size_t)8 * 16 * 4;   // 82,432 bytes
    hipLaunchKernelGGL(k_attn, dim3(L_LEN / 128, HEADS_), dim3(256), smem, stream,
                       q_bf, k_bf, v_bf, o_bf);
  }

  // 4) out projection + LN + residual
  hipLaunchKernelGGL(k_gemm, dim3(DIM_ / 64, L_LEN / 128), dim3(256), 0, stream,
                     o_bf, ow_bf, out_b, attn_f, (bf16*)nullptr, L_LEN, DIM_, DIM_);
  hipLaunchKernelGGL(k_ln, dim3(L_LEN / 8), dim3(256), 0, stream,
                     attn_f, (const float*)nullptr, residual, cnn_f, cnn_bf);

  // 5) conv stack (im2col GEMM + softplus) x3
  for (int l = 0; l < 3; l++) {
    const bf16* src_l = (l == 0) ? cnn_bf : act_bf;
    hipLaunchKernelGGL(k_im2col, dim3(cdiv(L_LEN * (KCONV / 8), 256)), dim3(256), 0, stream,
                       src_l, col_bf);
    hipLaunchKernelGGL(k_gemm, dim3(DIM_ / 64, L_LEN / 128), dim3(256), 0, stream,
                       col_bf, cw_bf + (size_t)l * DIM_ * KCONV, conv_b + l * DIM_,
                       h_f, (bf16*)nullptr, L_LEN, DIM_, KCONV);
    hipLaunchKernelGGL(k_softplus, dim3(cdiv(NT, 256)), dim3(256), 0, stream,
                       h_f, act_f, act_bf, NT);
  }

  // 6) skip GEMM + final LN -> d_out
  hipLaunchKernelGGL(k_gemm, dim3(DIM_ / 64, L_LEN / 128), dim3(256), 0, stream,
                     cnn_bf, sw_bf, skip_b, skip_f, (bf16*)nullptr, L_LEN, DIM_, DIM_);
  hipLaunchKernelGGL(k_ln, dim3(L_LEN / 8), dim3(256), 0, stream,
                     act_f, skip_f, (const float*)nullptr, outp, (bf16*)nullptr);
}
